// GATLayerMultihead_34291018891895
// MI455X (gfx1250) — matrix operand order repaired and checked by reading, not compiled, not run on hardware
//
#include <hip/hip_runtime.h>

typedef __attribute__((ext_vector_type(16))) _Float16 v16h;
typedef __attribute__((ext_vector_type(8)))  float    v8f;
typedef __attribute__((ext_vector_type(4)))  float    v4f;

#define NTOT 4096
#define FDIM 64
#define NH   8

// ---------------------------------------------------------------------------
// Phase 1: ltg[h,n,k] = sum_f graph[n,f] * W[h,f,k]   (written as f16 for WMMA B)
//          s[h,n] = ltg[h,n,:] . a[h,:64]
//          d[h,n] = ltg[h,n,:] . a[h,64:]
// grid = (N/64, H), block = 256. Each thread computes 16 outputs of one row.
// ---------------------------------------------------------------------------
__global__ __launch_bounds__(256)
void gat_phase1(const float* __restrict__ graph, const float* __restrict__ W,
                const float* __restrict__ a, _Float16* __restrict__ ltg_h,
                float* __restrict__ s_out, float* __restrict__ d_out)
{
    __shared__ float Wl[64 * 64];     // 16 KB: W for this head
    __shared__ float Gl[64 * 64];     // 16 KB: 64 graph rows
    __shared__ float red_s[64][4];
    __shared__ float red_d[64][4];

    const int h   = blockIdx.y;
    const int n0  = blockIdx.x * 64;
    const int tid = threadIdx.x;

    for (int i = tid; i < 64 * 64; i += 256) Wl[i] = W[(size_t)h * 64 * 64 + i];
    for (int i = tid; i < 64 * 64; i += 256) Gl[i] = graph[(size_t)n0 * 64 + i];
    __syncthreads();

    const int rl  = tid >> 2;   // local row 0..63
    const int seg = tid & 3;    // 16-col segment 0..3

    float acc[16];
#pragma unroll
    for (int c = 0; c < 16; ++c) acc[c] = 0.0f;

    for (int k = 0; k < 64; ++k) {
        const float g = Gl[rl * 64 + k];
        const float* wr = &Wl[k * 64 + seg * 16];
#pragma unroll
        for (int c = 0; c < 16; ++c) acc[c] = fmaf(g, wr[c], acc[c]);
    }

    const int row = n0 + rl;
    _Float16* dst = ltg_h + ((size_t)h * NTOT + row) * FDIM + seg * 16;
#pragma unroll
    for (int c = 0; c < 16; ++c) dst[c] = (_Float16)acc[c];

    // partial s / d reductions
    float ps = 0.0f, pd = 0.0f;
#pragma unroll
    for (int c = 0; c < 16; ++c) {
        ps = fmaf(acc[c], a[h * 2 * FDIM + seg * 16 + c], ps);
        pd = fmaf(acc[c], a[h * 2 * FDIM + FDIM + seg * 16 + c], pd);
    }
    red_s[rl][seg] = ps;
    red_d[rl][seg] = pd;
    __syncthreads();
    if (seg == 0) {
        s_out[(size_t)h * NTOT + row] = red_s[rl][0] + red_s[rl][1] + red_s[rl][2] + red_s[rl][3];
        d_out[(size_t)h * NTOT + row] = red_d[rl][0] + red_d[rl][1] + red_d[rl][2] + red_d[rl][3];
    }
}

// ---------------------------------------------------------------------------
// Phase 2: fused softmax-matmul, exact row max via monotonicity:
//   max_j lrelu(s_i + d_j) = lrelu(s_i + max_j d_j)  -> no online rescaling.
// Row sums computed by an extra WMMA against an all-ones B tile, landing in
// exactly the C layout needed for the final divide (no shuffles).
// grid = (N/32, H), block = 32 (one wave32). Each wave owns a 32x64 output
// tile (2 A tiles sharing one 32x64 B chunk): 10 WMMAs per j-chunk.
// ---------------------------------------------------------------------------
__global__ __launch_bounds__(32)
void gat_phase2(const _Float16* __restrict__ ltg_h,
                const float* __restrict__ s_vec, const float* __restrict__ d_vec,
                float* __restrict__ out)
{
    __shared__ float dK_lds[NTOT];           // 16 KB: LOG2E * d[h,:]

    const int h    = blockIdx.y;
    const int i0   = blockIdx.x * 32;
    const int lane = threadIdx.x;
    const float LOG2E = 1.4426950408889634f;

    // stage LOG2E*d[h,:] into LDS; track max(d) on the fly
    float dmax = -3.0e38f;
    {
        const v4f* src = (const v4f*)(d_vec + (size_t)h * NTOT);
        v4f* dst = (v4f*)dK_lds;
        for (int i = lane; i < NTOT / 4; i += 32) {
            v4f v = src[i];
            dmax = fmaxf(fmaxf(fmaxf(v.x, v.y), fmaxf(v.z, v.w)), dmax);
            v4f vk = { v.x * LOG2E, v.y * LOG2E, v.z * LOG2E, v.w * LOG2E };
            dst[i] = vk;
        }
#pragma unroll
        for (int off = 16; off >= 1; off >>= 1)
            dmax = fmaxf(dmax, __shfl_xor(dmax, off, 32));
    }
    __syncthreads();

    const int mrow  = lane & 15;             // A-matrix row owned by this lane
    const int rbase = (lane >> 4) << 3;      // C-matrix row base: 0 or 8
    const int cbase = (lane >> 4) << 3;      // A-layout K base: 0 or 8

    // per-tile row constants: Ks = LOG2E*s, Km = LOG2E*lrelu(s+dmax)
    float Ks[2], Km[2];
#pragma unroll
    for (int t = 0; t < 2; ++t) {
        const float s = s_vec[(size_t)h * NTOT + i0 + 16 * t + mrow];
        const float x = s + dmax;
        const float m = fmaxf(x, 0.2f * x); // lrelu (exact row max)
        Ks[t] = LOG2E * s;
        Km[t] = LOG2E * m;
    }

    v8f c0a = {}, c0b = {}, c0c = {}, c0d = {};   // tile 0: rows i0+0..15
    v8f c1a = {}, c1b = {}, c1c = {}, c1d = {};   // tile 1: rows i0+16..31
    v8f sum0 = {}, sum1 = {};                     // softmax denominators

    v16h bones;
#pragma unroll
    for (int i = 0; i < 16; ++i) bones[i] = (_Float16)1.0f;

    const _Float16* vbase = ltg_h + (size_t)h * NTOT * FDIM;

    for (int j0 = 0; j0 < NTOT; j0 += 32) {
        // --- B chunk: lane holds column N=lane%16 of each 32x16 tile, K in table order (cbase+{0..7}, 16+cbase+{0..7}) ---
        const _Float16* vcol = vbase + (size_t)j0 * FDIM + (lane & 15);
        v16h b0, b1, b2, b3;
#pragma unroll
        for (int s = 0; s < 16; ++s) {
            const int kk = (s < 8) ? (cbase + s) : (16 + cbase + (s - 8));
            const _Float16* p = vcol + (size_t)kk * FDIM;
            b0[s] = p[0];
            b1[s] = p[16];
            b2[s] = p[32];
            b3[s] = p[48];
        }
        if (j0 + 32 < NTOT)
            __builtin_prefetch(vbase + (size_t)(j0 + 32 + lane) * FDIM, 0, 1);

        // --- 16 scaled-d values for this lane's A columns (shared by both tiles)
        // cols = j0 + cbase + {0..7} and j0 + cbase + {16..23}
        v4f q0 = *(const v4f*)&dK_lds[j0 + cbase +  0];
        v4f q1 = *(const v4f*)&dK_lds[j0 + cbase +  4];
        v4f q2 = *(const v4f*)&dK_lds[j0 + cbase + 16];
        v4f q3 = *(const v4f*)&dK_lds[j0 + cbase + 20];
        float dv[16] = { q0.x, q0.y, q0.z, q0.w, q1.x, q1.y, q1.z, q1.w,
                         q2.x, q2.y, q2.z, q2.w, q3.x, q3.y, q3.z, q3.w };

        // --- P tiles: p = exp2( lrelu_scaled(Ks + Kd) - Km ) ---
        v16h A0 = {}, A1 = {};
#pragma unroll
        for (int i = 0; i < 16; ++i) {
            float x0 = Ks[0] + dv[i];
            float x1 = Ks[1] + dv[i];
            float y0 = fmaxf(x0, 0.2f * x0);
            float y1 = fmaxf(x1, 0.2f * x1);
            A0[i] = (_Float16)__builtin_amdgcn_exp2f(y0 - Km[0]);
            A1[i] = (_Float16)__builtin_amdgcn_exp2f(y1 - Km[1]);
        }

        // --- 10 WMMAs: 2x4 PV + 2 row-sum tiles ---
        sum0 = __builtin_amdgcn_wmma_f32_16x16x32_f16(false, A0, false, bones, (short)0, sum0, false, false);
        sum1 = __builtin_amdgcn_wmma_f32_16x16x32_f16(false, A1, false, bones, (short)0, sum1, false, false);
        c0a = __builtin_amdgcn_wmma_f32_16x16x32_f16(false, A0, false, b0, (short)0, c0a, false, false);
        c0b = __builtin_amdgcn_wmma_f32_16x16x32_f16(false, A0, false, b1, (short)0, c0b, false, false);
        c0c = __builtin_amdgcn_wmma_f32_16x16x32_f16(false, A0, false, b2, (short)0, c0c, false, false);
        c0d = __builtin_amdgcn_wmma_f32_16x16x32_f16(false, A0, false, b3, (short)0, c0d, false, false);
        c1a = __builtin_amdgcn_wmma_f32_16x16x32_f16(false, A1, false, b0, (short)0, c1a, false, false);
        c1b = __builtin_amdgcn_wmma_f32_16x16x32_f16(false, A1, false, b1, (short)0, c1b, false, false);
        c1c = __builtin_amdgcn_wmma_f32_16x16x32_f16(false, A1, false, b2, (short)0, c1c, false, false);
        c1d = __builtin_amdgcn_wmma_f32_16x16x32_f16(false, A1, false, b3, (short)0, c1d, false, false);
    }

    // --- epilogue: sum tile element (r, ncol) == rowsum(rbase+r): local divide ---
    float* obase = out + ((size_t)h * NTOT + i0) * FDIM;
    const int ncol = lane & 15;
#pragma unroll
    for (int r = 0; r < 8; ++r) {
        const float l0 = __builtin_amdgcn_rcpf(sum0[r]);
        const float l1 = __builtin_amdgcn_rcpf(sum1[r]);
        float* o0 = obase + (size_t)(rbase + r) * FDIM;
        float* o1 = obase + (size_t)(16 + rbase + r) * FDIM;
        o0[ 0 + ncol] = c0a[r] * l0;
        o0[16 + ncol] = c0b[r] * l0;
        o0[32 + ncol] = c0c[r] * l0;
        o0[48 + ncol] = c0d[r] * l0;
        o1[ 0 + ncol] = c1a[r] * l1;
        o1[16 + ncol] = c1b[r] * l1;
        o1[32 + ncol] = c1c[r] * l1;
        o1[48 + ncol] = c1d[r] * l1;
    }
}

// ---------------------------------------------------------------------------
extern "C" void kernel_launch(void* const* d_in, const int* in_sizes, int n_in,
                              void* d_out, int out_size, void* d_ws, size_t ws_size,
                              hipStream_t stream)
{
    const float* graph = (const float*)d_in[0];   // (N, 64) f32
    const float* W     = (const float*)d_in[1];   // (H, 64, 64) f32
    const float* a     = (const float*)d_in[2];   // (H, 128) f32
    float* out = (float*)d_out;                   // (H, N, 64) f32

    char* ws = (char*)d_ws;
    _Float16* ltg_h = (_Float16*)ws;                                   // 4 MB
    float* s_vec = (float*)(ws + (size_t)NH * NTOT * FDIM * sizeof(_Float16));
    float* d_vec = s_vec + (size_t)NH * NTOT;                          // 128 KB each

    dim3 g1(NTOT / 64, NH), b1(256);
    gat_phase1<<<g1, b1, 0, stream>>>(graph, W, a, ltg_h, s_vec, d_vec);

    dim3 g2(NTOT / 32, NH), b2(32);
    gat_phase2<<<g2, b2, 0, stream>>>(ltg_h, s_vec, d_vec, out);
}
